// TemporalEdgeGNN_77137612636694
// MI455X (gfx1250) — matrix operand order repaired and checked by reading, not compiled, not run on hardware
//
#include <hip/hip_runtime.h>
#include <hip/hip_bf16.h>

typedef __attribute__((ext_vector_type(16))) _Float16 v16h;
typedef __attribute__((ext_vector_type(8)))  _Float16 v8h;
typedef __attribute__((ext_vector_type(4)))  _Float16 v4h;
typedef __attribute__((ext_vector_type(8)))  float    v8f;

static inline int cdiv(int a, int b) { return (a + b - 1) / b; }

__device__ inline void atomicMaxF(float* addr, float val) {
  if (val >= 0.0f) atomicMax((int*)addr, __float_as_int(val));
  else             atomicMin((unsigned int*)addr, __float_as_uint(val));
}

// ---- fragment helpers -----------------------------------------------------
__device__ inline v8h cvt8(const float4 a, const float4 b) {
  v8h r;
  r[0] = (_Float16)a.x; r[1] = (_Float16)a.y; r[2] = (_Float16)a.z; r[3] = (_Float16)a.w;
  r[4] = (_Float16)b.x; r[5] = (_Float16)b.y; r[6] = (_Float16)b.z; r[7] = (_Float16)b.w;
  return r;
}
__device__ inline v4h cvt4(const float4 a) {
  v4h r;
  r[0] = (_Float16)a.x; r[1] = (_Float16)a.y; r[2] = (_Float16)a.z; r[3] = (_Float16)a.w;
  return r;
}
__device__ inline v16h cat16(v8h lo, v8h hi) {
  return __builtin_shufflevector(lo, hi, 0, 1, 2, 3, 4, 5, 6, 7, 8, 9, 10, 11, 12, 13, 14, 15);
}
// A fragment (16x32 f16, ISA layout) from a 32-float contiguous row (f32 -> f16)
__device__ inline v16h afrag_g(const float* bp, int lh) {
  const float4* p0 = (const float4*)(bp + 8 * lh);
  const float4* p1 = (const float4*)(bp + 16 + 8 * lh);
  return cat16(cvt8(p0[0], p0[1]), cvt8(p1[0], p1[1]));
}
// A fragment from a 32-half contiguous f16 row (no conversion, 2x b128)
__device__ inline v16h afrag_g16(const _Float16* bp, int lh) {
  v8h lo = *(const v8h*)(bp + 8 * lh);
  v8h hi = *(const v8h*)(bp + 16 + 8 * lh);
  return cat16(lo, hi);
}
// A fragment from LDS row (rowp 16B-aligned, row stride multiple of 16B)
__device__ inline v16h afrag_l(const _Float16* rowp, int koff, int lh) {
  v8h lo = *(const v8h*)(rowp + koff + 8 * lh);
  v8h hi = *(const v8h*)(rowp + koff + 16 + 8 * lh);
  return cat16(lo, hi);
}
// B fragment from LDS row: halves at koff + 8*lh and koff + 16 + 8*lh
__device__ inline v16h bfrag_l(const _Float16* rowp, int koff, int lh) {
  v8h lo = *(const v8h*)(rowp + koff + 8 * lh);
  v8h hi = *(const v8h*)(rowp + koff + 16 + 8 * lh);
  return cat16(lo, hi);
}

// ---------------------------------------------------------------------------
// WMMA GEMM: Y[M, NCT*16] = X[M,K] @ W[NCT*16, K]^T  (f32 in, OutT out)
// ---------------------------------------------------------------------------
template <int NCT, typename OutT>
__global__ void __launch_bounds__(128)
k_feat_gemm(const float* __restrict__ X, const float* __restrict__ W,
            OutT* __restrict__ Y, int M, int K)
{
  const int Nout = NCT * 16;
  __shared__ _Float16 ldsX[64][40];        // 40-half stride = 80B (16B multiple)
  __shared__ _Float16 ldsW[NCT * 16][40];
  const int w = threadIdx.x >> 5, lane = threadIdx.x & 31;
  const int lm = lane & 15, lh = lane >> 4;
  const int rowbase = blockIdx.x * 64;
  const v8f vzero = {};
  v8f acc[NCT];
#pragma unroll
  for (int ct = 0; ct < NCT; ++ct) acc[ct] = vzero;

  const int srow = threadIdx.x >> 1, shalf = threadIdx.x & 1;
  int gr = rowbase + srow; if (gr >= M) gr = M - 1;      // clamp: branchless staging
  const float* xrow = X + (size_t)gr * K + shalf * 16;

  for (int k0 = 0; k0 < K; k0 += 32) {
    __syncthreads();
    {
      const float4* xp = (const float4*)(xrow + k0);
      *(v8h*)&ldsX[srow][shalf * 16]     = cvt8(xp[0], xp[1]);
      *(v8h*)&ldsX[srow][shalf * 16 + 8] = cvt8(xp[2], xp[3]);
    }
    for (int i = threadIdx.x; i < Nout * 8; i += 128) {
      int n = i >> 3, part = i & 7;
      const float4 f = *(const float4*)&W[(size_t)n * K + k0 + part * 4];
      *(v4h*)&ldsW[n][part * 4] = cvt4(f);
    }
    __syncthreads();
    v16h a = afrag_l(&ldsX[w * 16 + lm][0], 0, lh);
#pragma unroll
    for (int ct = 0; ct < NCT; ++ct) {
      v16h b = bfrag_l(&ldsW[ct * 16 + lm][0], 0, lh);
      acc[ct] = __builtin_amdgcn_wmma_f32_16x16x32_f16(false, a, false, b, (short)0,
                                                       acc[ct], false, false);
    }
  }
#pragma unroll
  for (int ct = 0; ct < NCT; ++ct) {
#pragma unroll
    for (int r = 0; r < 8; ++r) {
      int g = rowbase + w * 16 + r + 8 * lh;
      if (g < M) Y[(size_t)g * Nout + ct * 16 + lm] = (OutT)acc[ct][r];
    }
  }
}

// ---------------------------------------------------------------------------
// GAT attention (C = 64 fixed; features stored f16)
// ---------------------------------------------------------------------------
__global__ void k_alpha(const _Float16* __restrict__ Hf, const float* __restrict__ as_,
                        const float* __restrict__ ad_, float* __restrict__ als,
                        float* __restrict__ ald, int n, int heads)
{
  int t = blockIdx.x * blockDim.x + threadIdx.x;
  if (t >= n * heads) return;
  int node = t / heads, hd = t % heads;
  const v8h* hp = (const v8h*)(Hf + (size_t)node * heads * 64 + hd * 64);
  const float* ap = as_ + hd * 64;
  const float* dp = ad_ + hd * 64;
  float s = 0.f, d = 0.f;
#pragma unroll
  for (int c = 0; c < 8; ++c) {
    v8h v = hp[c];
#pragma unroll
    for (int j = 0; j < 8; ++j) {
      float vf = (float)v[j];
      s += vf * ap[c * 8 + j];
      d += vf * dp[c * 8 + j];
    }
  }
  als[t] = s; ald[t] = d;
}

__global__ void k_init(float* __restrict__ segmax, float* __restrict__ segsum, int nh)
{
  int i = blockIdx.x * blockDim.x + threadIdx.x;
  if (i < nh) { segmax[i] = -1e30f; segsum[i] = 0.0f; }
}

__global__ void k_edge_max(const int* __restrict__ src, const int* __restrict__ dst,
                           const float* __restrict__ als, const float* __restrict__ ald,
                           float* __restrict__ segmax, int E, int n, int heads)
{
  int i = blockIdx.x * blockDim.x + threadIdx.x;
  if (i >= E + n) return;
  int s = (i < E) ? src[i] : (i - E);   // self-loops appended
  int d = (i < E) ? dst[i] : (i - E);
  for (int hd = 0; hd < heads; ++hd) {
    float a = als[s * heads + hd] + ald[d * heads + hd];
    a = (a > 0.f) ? a : 0.2f * a;       // leaky_relu 0.2
    atomicMaxF(&segmax[d * heads + hd], a);
  }
}

__global__ void k_edge_exp(const int* __restrict__ src, const int* __restrict__ dst,
                           const float* __restrict__ als, const float* __restrict__ ald,
                           const float* __restrict__ segmax, float* __restrict__ segsum,
                           float* __restrict__ expv, int E, int n, int heads)
{
  int i = blockIdx.x * blockDim.x + threadIdx.x;
  if (i >= E + n) return;
  int s = (i < E) ? src[i] : (i - E);
  int d = (i < E) ? dst[i] : (i - E);
  for (int hd = 0; hd < heads; ++hd) {
    float a = als[s * heads + hd] + ald[d * heads + hd];
    a = (a > 0.f) ? a : 0.2f * a;
    float e = expf(a - segmax[d * heads + hd]);
    expv[(size_t)i * heads + hd] = e;
    atomicAdd(&segsum[d * heads + hd], e);
  }
}

// ---- CSR build (dst + self-loops), shared by both GAT layers --------------
__global__ void k_zeroi(int* __restrict__ p, int n) {
  int i = blockIdx.x * blockDim.x + threadIdx.x;
  if (i < n) p[i] = 0;
}

__global__ void k_hist(const int* __restrict__ dst, int* __restrict__ counts, int E, int n)
{
  int i = blockIdx.x * blockDim.x + threadIdx.x;
  if (i >= E + n) return;
  int d = (i < E) ? dst[i] : (i - E);
  atomicAdd(&counts[d], 1);
}

// single-block chunked exclusive scan: rowptr[0..n], rowptr[n] = total
__global__ void __launch_bounds__(256)
k_scan(const int* __restrict__ counts, int* __restrict__ rowptr, int n)
{
  __shared__ int partial[256];
  int tid = threadIdx.x;
  int chunk = (n + 255) / 256;
  int lo = tid * chunk, hi = lo + chunk; if (hi > n) hi = n;
  int s = 0;
  for (int i = lo; i < hi; ++i) s += counts[i];
  partial[tid] = s;
  __syncthreads();
  for (int off = 1; off < 256; off <<= 1) {
    int v = (tid >= off) ? partial[tid - off] : 0;
    __syncthreads();
    partial[tid] += v;
    __syncthreads();
  }
  int run = (tid == 0) ? 0 : partial[tid - 1];
  for (int i = lo; i < hi; ++i) { rowptr[i] = run; run += counts[i]; }
  if (tid == 255) rowptr[n] = run;
}

__global__ void k_fill(const int* __restrict__ dst, const int* __restrict__ rowptr,
                       int* __restrict__ fill, int* __restrict__ eidx, int E, int n)
{
  int i = blockIdx.x * blockDim.x + threadIdx.x;
  if (i >= E + n) return;
  int d = (i < E) ? dst[i] : (i - E);
  int slot = rowptr[d] + atomicAdd(&fill[d], 1);
  eidx[slot] = i;
}

// ---- atomic-free attention aggregation + head-mean + bias + elu -----------
// thread = (node, c); one wave spans 32 channels of one node -> zero divergence
template <int HEADS, typename OutT>
__global__ void k_accum_csr(const int* __restrict__ rowptr, const int* __restrict__ eidx,
                            const int* __restrict__ src, const float* __restrict__ expv,
                            const float* __restrict__ segsum, const _Float16* __restrict__ Hf,
                            const float* __restrict__ bias, OutT* __restrict__ out,
                            int n, int E)
{
  int idx = blockIdx.x * blockDim.x + threadIdx.x;
  if (idx >= n * 64) return;
  int node = idx >> 6, c = idx & 63;
  int jb = rowptr[node], je = rowptr[node + 1];
  float a0 = 0.f, a1 = 0.f;
  for (int j = jb; j < je; ++j) {
    int i = eidx[j];
    int s = (i < E) ? src[i] : node;
    a0 += expv[(size_t)i * HEADS + 0] * (float)Hf[((size_t)s * HEADS + 0) * 64 + c];
    if (HEADS == 2)
      a1 += expv[(size_t)i * HEADS + 1] * (float)Hf[((size_t)s * HEADS + 1) * 64 + c];
  }
  float v;
  if (HEADS == 2) {
    v = 0.5f * (a0 / segsum[node * 2 + 0] + a1 / segsum[node * 2 + 1]);
  } else {
    v = a0 / segsum[node];
  }
  v += bias[c];
  out[idx] = (OutT)((v > 0.f) ? v : (expf(v) - 1.0f));   // elu
}

// ---------------------------------------------------------------------------
// Edge MLP (f16 gather + 160->64->64 WMMA GEMMs) + ctx partial sums
// ---------------------------------------------------------------------------
__global__ void __launch_bounds__(128)
k_edgemlp(const _Float16* __restrict__ h2, const int* __restrict__ src,
          const int* __restrict__ dst, const float* __restrict__ eat,
          const float* __restrict__ We1, const float* __restrict__ be1,
          const float* __restrict__ We2, const float* __restrict__ be2,
          float* __restrict__ ctx_sum, int E)
{
  __shared__ _Float16 w1s[64][160];       // 320B stride
  __shared__ _Float16 w2s[64][64];        // 128B stride
  __shared__ _Float16 ehs[4][16][72];     // 144B stride, per-wave scratch
  for (int i = threadIdx.x; i < 64 * 40; i += 128) {
    int n = i / 40, part = i % 40;
    *(v4h*)&w1s[n][part * 4] = cvt4(*(const float4*)&We1[n * 160 + part * 4]);
  }
  for (int i = threadIdx.x; i < 64 * 16; i += 128) {
    int n = i >> 4, part = i & 15;
    *(v4h*)&w2s[n][part * 4] = cvt4(*(const float4*)&We2[n * 64 + part * 4]);
  }
  __syncthreads();

  const int w = threadIdx.x >> 5, lane = threadIdx.x & 31;
  const int lm = lane & 15, lh = lane >> 4;
  const v8f vzero = {};
  const int ntiles = (E + 63) >> 6;

  for (int tb = blockIdx.x; tb < ntiles; tb += gridDim.x) {
    const int eb = (tb * 4 + w) * 16;
    const int me = eb + lm;
    const int mec = (me < E) ? me : (E - 1);   // clamp: invalid rows discarded later
    const int s = src[mec], d = dst[mec];
    const _Float16* h2s = h2 + (size_t)s * 64;
    const _Float16* h2d = h2 + (size_t)d * 64;
    const float*    ep  = eat + (size_t)mec * 32;
    const _Float16* basep16[4] = { h2s, h2s + 32, h2d, h2d + 32 };
    __builtin_prefetch(eat + ((size_t)eb + 64 * gridDim.x) * 32, 0, 0);

    v8f acc[4];
#pragma unroll
    for (int ct = 0; ct < 4; ++ct) acc[ct] = vzero;
#pragma unroll
    for (int kc = 0; kc < 4; ++kc) {           // h2[src] | h2[dst] chunks (f16 direct)
      v16h a = afrag_g16(basep16[kc], lh);
#pragma unroll
      for (int ct = 0; ct < 4; ++ct) {
        v16h b = bfrag_l(&w1s[ct * 16 + lm][0], kc * 32, lh);
        acc[ct] = __builtin_amdgcn_wmma_f32_16x16x32_f16(false, a, false, b, (short)0,
                                                         acc[ct], false, false);
      }
    }
    {                                          // edge_attr chunk (f32 -> f16)
      v16h a = afrag_g(ep, lh);
#pragma unroll
      for (int ct = 0; ct < 4; ++ct) {
        v16h b = bfrag_l(&w1s[ct * 16 + lm][0], 128, lh);
        acc[ct] = __builtin_amdgcn_wmma_f32_16x16x32_f16(false, a, false, b, (short)0,
                                                         acc[ct], false, false);
      }
    }
#pragma unroll
    for (int ct = 0; ct < 4; ++ct) {
      int n = ct * 16 + lm;
      float bn = be1[n];
#pragma unroll
      for (int r = 0; r < 8; ++r) {
        float v = acc[ct][r] + bn;
        ehs[w][r + 8 * lh][n] = (_Float16)(v > 0.f ? v : 0.f);
      }
    }
    v8f acc2[4];
#pragma unroll
    for (int ct = 0; ct < 4; ++ct) acc2[ct] = vzero;
#pragma unroll
    for (int kc = 0; kc < 2; ++kc) {
      v16h a = afrag_l(&ehs[w][lm][0], kc * 32, lh);
#pragma unroll
      for (int ct = 0; ct < 4; ++ct) {
        v16h b = bfrag_l(&w2s[ct * 16 + lm][0], kc * 32, lh);
        acc2[ct] = __builtin_amdgcn_wmma_f32_16x16x32_f16(false, a, false, b, (short)0,
                                                          acc2[ct], false, false);
      }
    }
#pragma unroll
    for (int ct = 0; ct < 4; ++ct) {
      int n = ct * 16 + lm;
      float bn = be2[n];
      float csum = 0.0f;
#pragma unroll
      for (int r = 0; r < 8; ++r) {
        float v = acc2[ct][r] + bn;
        v = v > 0.f ? v : 0.f;
        if (eb + r + 8 * lh < E) csum += v;
      }
      csum += __shfl_xor(csum, 16);
      if (lh == 0) atomicAdd(&ctx_sum[n], csum);
    }
  }
}

// ---------------------------------------------------------------------------
// Tiny LSTM over T=4 + temporal-context bias precompute (single block)
// ---------------------------------------------------------------------------
__global__ void k_lstm(const float* __restrict__ ctx_sum, const float* __restrict__ Wih,
                       const float* __restrict__ Whh, const float* __restrict__ bih,
                       const float* __restrict__ bhh, const float* __restrict__ Wc1,
                       const float* __restrict__ bc1, float* __restrict__ temporal,
                       float* __restrict__ tvec, float invE)
{
  __shared__ float hprev[64], cprev[64], ctxv[64], g[256];
  int tid = threadIdx.x;
  if (tid < 64) { hprev[tid] = 0.f; cprev[tid] = 0.f; }
  for (int t = 0; t < 4; ++t) {
    __syncthreads();
    if (tid < 64) ctxv[tid] = ctx_sum[t * 64 + tid] * invE;
    __syncthreads();
    float a = bih[tid] + bhh[tid];
    for (int k = 0; k < 64; ++k)
      a += Wih[tid * 64 + k] * ctxv[k] + Whh[tid * 64 + k] * hprev[k];
    g[tid] = a;
    __syncthreads();
    if (tid < 64) {
      float ig = 1.f / (1.f + expf(-g[tid]));
      float fg = 1.f / (1.f + expf(-g[64 + tid]));
      float gg = tanhf(g[128 + tid]);
      float og = 1.f / (1.f + expf(-g[192 + tid]));
      float c  = fg * cprev[tid] + ig * gg;
      float hh = og * tanhf(c);
      cprev[tid] = c; hprev[tid] = hh;
      temporal[t * 64 + tid] = hh;
    }
    __syncthreads();
    if (tid < 64) {
      float a2 = bc1[tid];
      for (int k = 0; k < 64; ++k)
        a2 += Wc1[tid * 128 + 64 + k] * hprev[k];
      tvec[t * 64 + tid] = a2;     // Wc1[:,64:]@h_t + bc1
    }
  }
}

// ---------------------------------------------------------------------------
// Classifier: recompute eh, third WMMA GEMM (+tvec bias), relu, dot Wc2
// ---------------------------------------------------------------------------
__global__ void __launch_bounds__(128)
k_classify(const _Float16* __restrict__ h2, const int* __restrict__ src,
           const int* __restrict__ dst, const float* __restrict__ eat,
           const float* __restrict__ We1, const float* __restrict__ be1,
           const float* __restrict__ We2, const float* __restrict__ be2,
           const float* __restrict__ Wc1, const float* __restrict__ tvec_t,
           const float* __restrict__ Wc2, const float* __restrict__ bc2,
           float* __restrict__ out_t, int E)
{
  __shared__ _Float16 w1s[64][160];
  __shared__ _Float16 w2s[64][64];
  __shared__ _Float16 wcs[64][64];
  __shared__ _Float16 ehs[4][16][72];
  for (int i = threadIdx.x; i < 64 * 40; i += 128) {
    int n = i / 40, part = i % 40;
    *(v4h*)&w1s[n][part * 4] = cvt4(*(const float4*)&We1[n * 160 + part * 4]);
  }
  for (int i = threadIdx.x; i < 64 * 16; i += 128) {
    int n = i >> 4, part = i & 15;
    *(v4h*)&w2s[n][part * 4] = cvt4(*(const float4*)&We2[n * 64 + part * 4]);
    *(v4h*)&wcs[n][part * 4] = cvt4(*(const float4*)&Wc1[(size_t)n * 128 + part * 4]);
  }
  __syncthreads();

  const int w = threadIdx.x >> 5, lane = threadIdx.x & 31;
  const int lm = lane & 15, lh = lane >> 4;
  const v8f vzero = {};
  const float bc2v = bc2[0];
  const int ntiles = (E + 63) >> 6;

  for (int tb = blockIdx.x; tb < ntiles; tb += gridDim.x) {
    const int eb = (tb * 4 + w) * 16;
    const int me = eb + lm;
    const int mec = (me < E) ? me : (E - 1);
    const int s = src[mec], d = dst[mec];
    const _Float16* h2s = h2 + (size_t)s * 64;
    const _Float16* h2d = h2 + (size_t)d * 64;
    const float*    ep  = eat + (size_t)mec * 32;
    const _Float16* basep16[4] = { h2s, h2s + 32, h2d, h2d + 32 };
    __builtin_prefetch(eat + ((size_t)eb + 64 * gridDim.x) * 32, 0, 0);

    v8f acc[4];
#pragma unroll
    for (int ct = 0; ct < 4; ++ct) acc[ct] = vzero;
#pragma unroll
    for (int kc = 0; kc < 4; ++kc) {
      v16h a = afrag_g16(basep16[kc], lh);
#pragma unroll
      for (int ct = 0; ct < 4; ++ct) {
        v16h b = bfrag_l(&w1s[ct * 16 + lm][0], kc * 32, lh);
        acc[ct] = __builtin_amdgcn_wmma_f32_16x16x32_f16(false, a, false, b, (short)0,
                                                         acc[ct], false, false);
      }
    }
    {
      v16h a = afrag_g(ep, lh);
#pragma unroll
      for (int ct = 0; ct < 4; ++ct) {
        v16h b = bfrag_l(&w1s[ct * 16 + lm][0], 128, lh);
        acc[ct] = __builtin_amdgcn_wmma_f32_16x16x32_f16(false, a, false, b, (short)0,
                                                         acc[ct], false, false);
      }
    }
#pragma unroll
    for (int ct = 0; ct < 4; ++ct) {
      int n = ct * 16 + lm;
      float bn = be1[n];
#pragma unroll
      for (int r = 0; r < 8; ++r) {
        float v = acc[ct][r] + bn;
        ehs[w][r + 8 * lh][n] = (_Float16)(v > 0.f ? v : 0.f);
      }
    }
    v8f acc2[4];
#pragma unroll
    for (int ct = 0; ct < 4; ++ct) acc2[ct] = vzero;
#pragma unroll
    for (int kc = 0; kc < 2; ++kc) {
      v16h a = afrag_l(&ehs[w][lm][0], kc * 32, lh);
#pragma unroll
      for (int ct = 0; ct < 4; ++ct) {
        v16h b = bfrag_l(&w2s[ct * 16 + lm][0], kc * 32, lh);
        acc2[ct] = __builtin_amdgcn_wmma_f32_16x16x32_f16(false, a, false, b, (short)0,
                                                          acc2[ct], false, false);
      }
    }
    // eh = relu(acc2 + be2) -> overwrite per-wave LDS tile (same wave, in-order)
#pragma unroll
    for (int ct = 0; ct < 4; ++ct) {
      int n = ct * 16 + lm;
      float bn = be2[n];
#pragma unroll
      for (int r = 0; r < 8; ++r) {
        float v = acc2[ct][r] + bn;
        ehs[w][r + 8 * lh][n] = (_Float16)(v > 0.f ? v : 0.f);
      }
    }
    // GEMM3: eh(64) @ Wc1[:, :64]^T + tvec -> relu -> dot Wc2
    v8f acc3[4];
#pragma unroll
    for (int ct = 0; ct < 4; ++ct) acc3[ct] = vzero;
#pragma unroll
    for (int kc = 0; kc < 2; ++kc) {
      v16h a = afrag_l(&ehs[w][lm][0], kc * 32, lh);
#pragma unroll
      for (int ct = 0; ct < 4; ++ct) {
        v16h b = bfrag_l(&wcs[ct * 16 + lm][0], kc * 32, lh);
        acc3[ct] = __builtin_amdgcn_wmma_f32_16x16x32_f16(false, a, false, b, (short)0,
                                                          acc3[ct], false, false);
      }
    }
#pragma unroll
    for (int r = 0; r < 8; ++r) {
      float p = 0.0f;
#pragma unroll
      for (int ct = 0; ct < 4; ++ct) {
        int n = ct * 16 + lm;
        float z = acc3[ct][r] + tvec_t[n];
        z = z > 0.f ? z : 0.f;
        p += z * Wc2[n];
      }
      p += __shfl_xor(p, 1);
      p += __shfl_xor(p, 2);
      p += __shfl_xor(p, 4);
      p += __shfl_xor(p, 8);
      int m = eb + r + 8 * lh;
      if (lm == 0 && m < E) out_t[m] = p + bc2v;
    }
  }
}

__global__ void k_zero(float* __restrict__ pz, int n) {
  int i = blockIdx.x * blockDim.x + threadIdx.x;
  if (i < n) pz[i] = 0.f;
}

// ---------------------------------------------------------------------------
extern "C" void kernel_launch(void* const* d_in, const int* in_sizes, int n_in,
                              void* d_out, int out_size, void* d_ws, size_t ws_size,
                              hipStream_t stream) {
  const float* x   = (const float*)d_in[0];
  const int*   ei  = (const int*)  d_in[1];
  const float* ea  = (const float*)d_in[2];
  const float* W1  = (const float*)d_in[3];
  const float* as1 = (const float*)d_in[4];
  const float* ad1 = (const float*)d_in[5];
  const float* b1  = (const float*)d_in[6];
  const float* W2  = (const float*)d_in[7];
  const float* as2 = (const float*)d_in[8];
  const float* ad2 = (const float*)d_in[9];
  const float* b2  = (const float*)d_in[10];
  const float* We1 = (const float*)d_in[11];
  const float* be1 = (const float*)d_in[12];
  const float* We2 = (const float*)d_in[13];
  const float* be2 = (const float*)d_in[14];
  const float* Wih = (const float*)d_in[15];
  const float* Whh = (const float*)d_in[16];
  const float* bih = (const float*)d_in[17];
  const float* bhh = (const float*)d_in[18];
  const float* Wc1 = (const float*)d_in[19];
  const float* bc1 = (const float*)d_in[20];
  const float* Wc2 = (const float*)d_in[21];
  const float* bc2 = (const float*)d_in[22];
  float* out = (float*)d_out;

  const int T = 4, N = 50000, E = 400000;
  const int PERSIST = 1024;   // persistent blocks for edge kernels

  // workspace carve-out (~59 MB total)
  char* p = (char*)d_ws;
  auto alloc = [&](size_t bytes) -> void* {
    void* r = (void*)p;
    p += (bytes + 255) & ~(size_t)255;
    return r;
  };
  _Float16* h2_all  = (_Float16*)alloc((size_t)T * N * 64 * 2);   // 25.6 MB (f16)
  _Float16* hfeat   = (_Float16*)alloc((size_t)N * 128 * 2);      // 12.8 MB (f16)
  float* h1buf      = (float*)alloc((size_t)N * 64 * 4);          // 12.8 MB
  float* als        = (float*)alloc((size_t)N * 2 * 4);
  float* ald        = (float*)alloc((size_t)N * 2 * 4);
  float* segmax     = (float*)alloc((size_t)N * 2 * 4);
  float* segsum     = (float*)alloc((size_t)N * 2 * 4);
  float* expv       = (float*)alloc((size_t)(E + N) * 2 * 4);     // 3.6 MB
  float* ctxs       = (float*)alloc((size_t)T * 64 * 4);
  float* temporal   = (float*)alloc((size_t)T * 64 * 4);
  float* tvec       = (float*)alloc((size_t)T * 64 * 4);
  int* counts       = (int*)alloc((size_t)N * 4);
  int* fill         = (int*)alloc((size_t)N * 4);
  int* rowptr       = (int*)alloc((size_t)(N + 1) * 4);
  int* eidx         = (int*)alloc((size_t)(E + N) * 4);           // 1.8 MB

  k_zero<<<1, 256, 0, stream>>>(ctxs, T * 64);

  for (int t = 0; t < T; ++t) {
    const float* xt   = x  + (size_t)t * N * 128;
    const int*   srcp = ei + (size_t)t * 2 * E;
    const int*   dstp = srcp + E;
    const float* eat  = ea + (size_t)t * E * 32;
    _Float16* h2t = h2_all + (size_t)t * N * 64;

    // CSR over dst (+self-loops), shared by both GAT layers of this snapshot
    k_zeroi<<<cdiv(N, 256), 256, 0, stream>>>(counts, N);
    k_zeroi<<<cdiv(N, 256), 256, 0, stream>>>(fill, N);
    k_hist<<<cdiv(E + N, 256), 256, 0, stream>>>(dstp, counts, E, N);
    k_scan<<<1, 256, 0, stream>>>(counts, rowptr, N);
    k_fill<<<cdiv(E + N, 256), 256, 0, stream>>>(dstp, rowptr, fill, eidx, E, N);

    // GAT layer 1 (heads = 2)
    k_feat_gemm<8, _Float16><<<cdiv(N, 64), 128, 0, stream>>>(xt, W1, hfeat, N, 128);
    k_alpha<<<cdiv(N * 2, 256), 256, 0, stream>>>(hfeat, as1, ad1, als, ald, N, 2);
    k_init<<<cdiv(N * 2, 256), 256, 0, stream>>>(segmax, segsum, N * 2);
    k_edge_max<<<cdiv(E + N, 256), 256, 0, stream>>>(srcp, dstp, als, ald, segmax, E, N, 2);
    k_edge_exp<<<cdiv(E + N, 256), 256, 0, stream>>>(srcp, dstp, als, ald, segmax, segsum, expv, E, N, 2);
    k_accum_csr<2, float><<<cdiv(N * 64, 256), 256, 0, stream>>>(rowptr, eidx, srcp, expv,
                                                                 segsum, hfeat, b1, h1buf, N, E);

    // GAT layer 2 (heads = 1)
    k_feat_gemm<4, _Float16><<<cdiv(N, 64), 128, 0, stream>>>(h1buf, W2, hfeat, N, 64);
    k_alpha<<<cdiv(N, 256), 256, 0, stream>>>(hfeat, as2, ad2, als, ald, N, 1);
    k_init<<<cdiv(N, 256), 256, 0, stream>>>(segmax, segsum, N);
    k_edge_max<<<cdiv(E + N, 256), 256, 0, stream>>>(srcp, dstp, als, ald, segmax, E, N, 1);
    k_edge_exp<<<cdiv(E + N, 256), 256, 0, stream>>>(srcp, dstp, als, ald, segmax, segsum, expv, E, N, 1);
    k_accum_csr<1, _Float16><<<cdiv(N * 64, 256), 256, 0, stream>>>(rowptr, eidx, srcp, expv,
                                                                    segsum, hfeat, b2, h2t, N, E);

    // Edge MLP + ctx partial sums (persistent)
    k_edgemlp<<<PERSIST, 128, 0, stream>>>(h2t, srcp, dstp, eat, We1, be1, We2, be2,
                                           ctxs + t * 64, E);
  }

  k_lstm<<<1, 256, 0, stream>>>(ctxs, Wih, Whh, bih, bhh, Wc1, bc1, temporal, tvec,
                                1.0f / (float)E);

  for (int t = 0; t < T; ++t) {
    const int*   srcp = ei + (size_t)t * 2 * E;
    const int*   dstp = srcp + E;
    const float* eat  = ea + (size_t)t * E * 32;
    const _Float16* h2t = h2_all + (size_t)t * N * 64;
    k_classify<<<PERSIST, 128, 0, stream>>>(h2t, srcp, dstp, eat, We1, be1, We2, be2,
                                            Wc1, tvec + t * 64, Wc2, bc2,
                                            out + (size_t)t * E, E);
  }

  (void)in_sizes; (void)n_in; (void)out_size; (void)ws_size;
}
